// PointNet2StagePointAttention_49151605735493
// MI455X (gfx1250) — matrix operand order repaired and checked by reading, not compiled, not run on hardware
//
#include <hip/hip_runtime.h>
#include <hip/hip_bf16.h>

typedef __attribute__((ext_vector_type(16))) __bf16 v16bf;
typedef __attribute__((ext_vector_type(8)))  __bf16 v8bf;
typedef __attribute__((ext_vector_type(8)))  float  v8f;
typedef __attribute__((ext_vector_type(4)))  int    v4i;

// ---------------------------------------------------------------------------
// gfx1250 async global->LDS path (guarded; falls back to sync copies)
// builtin signature: (v4i AS1* gsrc, v4i AS3* ldst, imm offset, imm cpol)
// ---------------------------------------------------------------------------
#if __has_builtin(__builtin_amdgcn_global_load_async_to_lds_b128) && \
    __has_builtin(__builtin_amdgcn_s_wait_asynccnt)
#define HAVE_ASYNC 1
#else
#define HAVE_ASYNC 0
#endif

typedef __attribute__((address_space(1))) v4i as1_v4i;
typedef __attribute__((address_space(3))) v4i as3_v4i;

__device__ __forceinline__ void async_copy16(void* lds_dst, const void* g_src)
{
#if HAVE_ASYNC
    __builtin_amdgcn_global_load_async_to_lds_b128(
        (as1_v4i*)(unsigned long long)g_src,
        (as3_v4i*)(unsigned long long)lds_dst, 0, 0);
#else
    *(uint4*)lds_dst = *(const uint4*)g_src;
#endif
}

__device__ __forceinline__ void async_fence_all()
{
#if HAVE_ASYNC
    __builtin_amdgcn_s_wait_asynccnt(0);
#endif
}

// ---------------------------------------------------------------------------
// Stage 1: per-point MLP  (xyz/rgb -> fx,fr -> gates -> pf2[128])
// writes pf2 as bf16, and atomic-max (float-as-int, all values >= 0) into vox1
// ---------------------------------------------------------------------------
__global__ __launch_bounds__(256)
void k_point_stage1(const float* __restrict__ inp, const int* __restrict__ idx,
                    const float* __restrict__ w1x, const float* __restrict__ b1x,
                    const float* __restrict__ w2x, const float* __restrict__ b2x,
                    const float* __restrict__ w1r, const float* __restrict__ b1r,
                    const float* __restrict__ w2r, const float* __restrict__ b2r,
                    const float* __restrict__ wax, const float* __restrict__ bax,
                    const float* __restrict__ war, const float* __restrict__ bar,
                    __bf16* __restrict__ pf2g, int* __restrict__ vox1, int n)
{
    __shared__ float s_w1x[192], s_b1x[64], s_w2x[2048], s_b2x[32];
    __shared__ float s_w1r[192], s_b1r[64], s_w2r[2048], s_b2r[32];
    __shared__ float s_wax[2048], s_bax[32], s_war[2048], s_bar[32];
    const int t = threadIdx.x;
    for (int j = t; j < 192; j += 256) { s_w1x[j] = w1x[j]; s_w1r[j] = w1r[j]; }
    for (int j = t; j < 64;  j += 256) { s_b1x[j] = b1x[j]; s_b1r[j] = b1r[j]; }
    for (int j = t; j < 2048; j += 256) {
        s_w2x[j] = w2x[j]; s_w2r[j] = w2r[j]; s_wax[j] = wax[j]; s_war[j] = war[j];
    }
    for (int j = t; j < 32; j += 256) {
        s_b2x[j] = b2x[j]; s_b2r[j] = b2r[j]; s_bax[j] = bax[j]; s_bar[j] = bar[j];
    }
    __syncthreads();

    const int i = blockIdx.x * 256 + t;
    const bool valid = i < n;
    const int ii = valid ? i : (n - 1);
    const float in0 = inp[(size_t)ii*6+0], in1 = inp[(size_t)ii*6+1], in2 = inp[(size_t)ii*6+2];
    const float in3 = inp[(size_t)ii*6+3], in4 = inp[(size_t)ii*6+4], in5 = inp[(size_t)ii*6+5];

    float h[64];
#pragma unroll
    for (int j = 0; j < 64; ++j)
        h[j] = fmaxf(in0*s_w1x[j] + in1*s_w1x[64+j] + in2*s_w1x[128+j] + s_b1x[j], 0.f);
    float fx[32];
#pragma unroll
    for (int j = 0; j < 32; ++j) {
        float a = s_b2x[j];
#pragma unroll
        for (int k = 0; k < 64; ++k) a += h[k] * s_w2x[k*32 + j];
        fx[j] = fmaxf(a, 0.f);
    }
#pragma unroll
    for (int j = 0; j < 64; ++j)
        h[j] = fmaxf(in3*s_w1r[j] + in4*s_w1r[64+j] + in5*s_w1r[128+j] + s_b1r[j], 0.f);
    float fr[32];
#pragma unroll
    for (int j = 0; j < 32; ++j) {
        float a = s_b2r[j];
#pragma unroll
        for (int k = 0; k < 64; ++k) a += h[k] * s_w2r[k*32 + j];
        fr[j] = fmaxf(a, 0.f);
    }
    float gx[32], gr[32];
#pragma unroll
    for (int j = 0; j < 32; ++j) {
        float ax = s_bax[j], ar = s_bar[j];
#pragma unroll
        for (int k = 0; k < 32; ++k) {
            ax += fx[k]*s_wax[k*32+j] + fr[k]*s_wax[(k+32)*32+j];
            ar += fx[k]*s_war[k*32+j] + fr[k]*s_war[(k+32)*32+j];
        }
        gx[j] = 1.f / (1.f + __expf(-ax));
        gr[j] = 1.f / (1.f + __expf(-ar));
    }
    if (!valid) return;
    const int vi = idx[i];
    __bf16* orow = pf2g + (size_t)i * 128;
    int* vrow = vox1 + (size_t)vi * 128;
    union { uint4 q[4]; __bf16 b[32]; } u;
#pragma unroll
    for (int j = 0; j < 32; ++j) { u.b[j] = (__bf16)fx[j]; }
#pragma unroll
    for (int j = 0; j < 4; ++j) ((uint4*)orow)[j] = u.q[j];
#pragma unroll
    for (int j = 0; j < 32; ++j) atomicMax(vrow + j, __float_as_int(fx[j]));
#pragma unroll
    for (int j = 0; j < 32; ++j) {
        float v = fx[j]*gx[j]; u.b[j] = (__bf16)v;
        atomicMax(vrow + 32 + j, __float_as_int(v));
    }
#pragma unroll
    for (int j = 0; j < 4; ++j) ((uint4*)(orow + 32))[j] = u.q[j];
#pragma unroll
    for (int j = 0; j < 32; ++j) { u.b[j] = (__bf16)fr[j]; }
#pragma unroll
    for (int j = 0; j < 4; ++j) ((uint4*)(orow + 64))[j] = u.q[j];
#pragma unroll
    for (int j = 0; j < 32; ++j) atomicMax(vrow + 64 + j, __float_as_int(fr[j]));
#pragma unroll
    for (int j = 0; j < 32; ++j) {
        float v = fr[j]*gr[j]; u.b[j] = (__bf16)v;
        atomicMax(vrow + 96 + j, __float_as_int(v));
    }
#pragma unroll
    for (int j = 0; j < 4; ++j) ((uint4*)(orow + 96))[j] = u.q[j];
}

// ---------------------------------------------------------------------------
// Pack weight (K x NC, row-major f32) into WMMA B-fragment layout (bf16):
// tile = nt*(K/32)+kt ; within tile, lane<16: N=lane, K=k0..k0+7 and k0+16..k0+23,
// lane>=16: N=lane-16, K=k0+8..k0+15 and k0+24..k0+31. Each lane owns 16 contiguous bf16.
// ---------------------------------------------------------------------------
__global__ __launch_bounds__(256)
void k_pack(const float* __restrict__ w, __bf16* __restrict__ dst, int K, int NC)
{
    const int gid = blockIdx.x * 256 + threadIdx.x;
    const int total = (K/32) * (NC/16) * 32;
    if (gid >= total) return;
    const int lane = gid & 31;
    const int tile = gid >> 5;
    const int KT = K / 32;
    const int kt = tile % KT, nt = tile / KT;
    const int nn = nt*16 + (lane & 15);
    const int k0 = kt*32 + (lane >> 4)*8;
    __bf16* d = dst + (size_t)tile*512 + lane*16;
#pragma unroll
    for (int e = 0; e < 16; ++e) d[e] = (__bf16)w[(size_t)(k0 + (e < 8 ? e : e + 8))*NC + nn];
}

// ---------------------------------------------------------------------------
// Per-wave 16-row x (TPW*16)-col WMMA tile over K; A staged in LDS (bf16),
// W fragments from either global or LDS (packed fragment-major)
// ---------------------------------------------------------------------------
template<int K, int TPW>
__device__ __forceinline__ void wmma_rowtile(const __bf16* As, const __bf16* Wp,
                                             v8f acc[TPW], int rg, int ns,
                                             int m, int hi, int lane)
{
    constexpr int KT = K / 32;
#pragma unroll
    for (int kt = 0; kt < KT; ++kt) {
        union { v16bf v; v8bf h[2]; } a;
        const int ab = (rg*16 + m)*K + kt*32 + hi*8;
        a.h[0] = *(const v8bf*)(As + ab);
        a.h[1] = *(const v8bf*)(As + ab + 16);
#pragma unroll
        for (int nt = 0; nt < TPW; ++nt) {
            const int gnt = ns*TPW + nt;
            const v16bf b = *(const v16bf*)(Wp + ((size_t)(gnt*KT + kt)*32 + lane)*16);
            acc[nt] = __builtin_amdgcn_wmma_f32_16x16x32_bf16(
                false, a.v, false, b, (short)0, acc[nt], false, false);
        }
    }
}

// ---------------------------------------------------------------------------
// Row-block GEMM: out = relu(A @ W + bias). A is f32 (rows x K) in global,
// W packed bf16 in global. 256 threads = 8 waves, 32 rows per block.
// Used for the two voxel-side GEMMs (wv1: 50k x 128, wv2: 50k x 256).
// ---------------------------------------------------------------------------
template<int K, int NC, bool OUTBF>
__global__ __launch_bounds__(256)
void k_gemm_rows(const float* __restrict__ Af, const __bf16* __restrict__ Wp,
                 const float* __restrict__ bias, void* __restrict__ outp, int rows)
{
    constexpr int TPW = NC / 64;
    __shared__ __align__(16) __bf16 As[32 * K];
    const int t = threadIdx.x;
    const int base = blockIdx.x * 32;
    for (int i = t; i < 32*K/4; i += 256) {
        const int row = (i*4) / K, c = (i*4) % K;
        const int gr = min(base + row, rows - 1);
        const float4 f = *(const float4*)(Af + (size_t)gr*K + c);
        __bf16* d = As + row*K + c;
        d[0] = (__bf16)f.x; d[1] = (__bf16)f.y; d[2] = (__bf16)f.z; d[3] = (__bf16)f.w;
    }
    __syncthreads();
    const int wid = t >> 5, lane = t & 31;
    const int rg = wid >> 2, ns = wid & 3;
    const int m = lane & 15, hi = lane >> 4;
    v8f acc[TPW] = {};
    wmma_rowtile<K, TPW>(As, Wp, acc, rg, ns, m, hi, lane);
#pragma unroll
    for (int nt = 0; nt < TPW; ++nt) {
        const int nn = ns*(NC/4) + nt*16 + m;
        const float bn = bias[nn];
#pragma unroll
        for (int r = 0; r < 8; ++r) {
            const int row = rg*16 + hi*8 + r;
            if (base + row < rows) {
                const float v = fmaxf(acc[nt][r] + bn, 0.f);
                if (OUTBF) ((__bf16*)outp)[(size_t)(base+row)*NC + nn] = (__bf16)v;
                else       ((float*)outp)[(size_t)(base+row)*NC + nn] = v;
            }
        }
    }
}

// ---------------------------------------------------------------------------
// Stage 2 (persistent): w3 + w4 packed bf16 resident in LDS (256 KB),
// double-buffered 32-row A tiles async-prefetched global->LDS.
// A = [voxA[idx] | pf2] (K=256); h = relu(A@w3+b3) staged in-place;
// pf5 = relu(h@w4+b4) scattered via atomic-max into vox2.
// LDS: 2*65536 (weights) + 2*32*256 (A ping-pong) bf16 = 288 KB (<320 KB/WGP)
// ---------------------------------------------------------------------------
#define S2_SMEM_BYTES ((2*65536 + 2*32*256) * 2)

__global__ __launch_bounds__(256)
void k_stage2_persist(const __bf16* __restrict__ pf2g, const __bf16* __restrict__ voxA,
                      const int* __restrict__ idx,
                      const __bf16* __restrict__ W3g, const float* __restrict__ b3,
                      const __bf16* __restrict__ W4g, const float* __restrict__ b4,
                      int* __restrict__ vox2, int n)
{
    extern __shared__ __align__(16) char smem[];
    __bf16* W3s = (__bf16*)smem;
    __bf16* W4s = W3s + 65536;
    __bf16* Ab0 = W4s + 65536;
    __bf16* Ab1 = Ab0 + 32*256;

    const int t = threadIdx.x;
    // Resident weight load: 256 KB, async global->LDS, 16B per op.
    for (int i = t*8; i < 65536; i += 256*8) {
        async_copy16(W3s + i, W3g + i);
        async_copy16(W4s + i, W4g + i);
    }

    const int nChunks = (n + 31) >> 5;
    const int row = t >> 3, seg = t & 7;

    const int wid = t >> 5, lane = t & 31;
    const int rg = wid >> 2, ns = wid & 3;
    const int m = lane & 15, hi = lane >> 4;

    // Stage one 32-row A tile (voxel-gather half + pf2 half) into LDS.
    auto stageA = [&](int chunk, __bf16* dst) {
        const int gbase = chunk * 32;
        const int gr = min(gbase + row, n - 1);
        const int vi = idx[gr];
        const __bf16* src = (seg < 4)
            ? (voxA + (size_t)vi*128 + seg*32)
            : (pf2g + (size_t)gr*128 + (seg-4)*32);
        __bf16* d = dst + row*256 + seg*32;
#pragma unroll
        for (int j = 0; j < 4; ++j) async_copy16(d + j*8, src + j*8);
    };

    int c = blockIdx.x;
    if (c < nChunks) stageA(c, Ab0);
    int p = 0;
    for (; c < nChunks; c += gridDim.x, p ^= 1) {
        async_fence_all();     // this wave's async writes to LDS landed
        __syncthreads();       // ... collectively (and prev chunk fully done)
        __bf16* A = p ? Ab1 : Ab0;
        __bf16* Anext = p ? Ab0 : Ab1;
        const int cn = c + gridDim.x;
        if (cn < nChunks) stageA(cn, Anext);   // overlap with compute below

        v8f acc[4] = {};
        wmma_rowtile<256, 4>(A, W3s, acc, rg, ns, m, hi, lane);
        __syncthreads();       // all waves done reading A
#pragma unroll
        for (int nt = 0; nt < 4; ++nt) {
            const int nn = ns*64 + nt*16 + m;
            const float bn = b3[nn];
#pragma unroll
            for (int r = 0; r < 8; ++r) {
                const int rr = rg*16 + hi*8 + r;
                A[rr*256 + nn] = (__bf16)fmaxf(acc[nt][r] + bn, 0.f);  // h in-place
            }
        }
        __syncthreads();       // h fully staged
        v8f acc2[4] = {};
        wmma_rowtile<256, 4>(A, W4s, acc2, rg, ns, m, hi, lane);

        const int gbase = c * 32;
#pragma unroll
        for (int nt = 0; nt < 4; ++nt) {
            const int nn = ns*64 + nt*16 + m;
            const float bn = b4[nn];
#pragma unroll
            for (int r = 0; r < 8; ++r) {
                const int grow = gbase + rg*16 + hi*8 + r;
                if (grow < n) {
                    const float v = fmaxf(acc2[nt][r] + bn, 0.f);
                    atomicMax(vox2 + (size_t)idx[grow]*256 + nn, __float_as_int(v));
                }
            }
        }
    }
}

// ---------------------------------------------------------------------------
extern "C" void kernel_launch(void* const* d_in, const int* in_sizes, int n_in,
                              void* d_out, int out_size, void* d_ws, size_t ws_size,
                              hipStream_t stream)
{
    const float* inp = (const float*)d_in[0];
    const int*   idx = (const int*)d_in[1];
    const int N = in_sizes[1];
    const int V = 50000;  // num_vox: fixed by reference
    const float* w1x = (const float*)d_in[3];  const float* b1x = (const float*)d_in[4];
    const float* w2x = (const float*)d_in[5];  const float* b2x = (const float*)d_in[6];
    const float* w1r = (const float*)d_in[7];  const float* b1r = (const float*)d_in[8];
    const float* w2r = (const float*)d_in[9];  const float* b2r = (const float*)d_in[10];
    const float* wax = (const float*)d_in[11]; const float* bax = (const float*)d_in[12];
    const float* war = (const float*)d_in[13]; const float* bar = (const float*)d_in[14];
    const float* wv1 = (const float*)d_in[15]; const float* bv1 = (const float*)d_in[16];
    const float* w3  = (const float*)d_in[17]; const float* b3  = (const float*)d_in[18];
    const float* w4  = (const float*)d_in[19]; const float* b4  = (const float*)d_in[20];
    const float* wv2 = (const float*)d_in[21]; const float* bv2 = (const float*)d_in[22];

    char* ws = (char*)d_ws;
    size_t off = 0;
    auto carve = [&](size_t bytes) -> void* {
        void* p = ws + off;
        off = (off + bytes + 255) & ~(size_t)255;
        return p;
    };
    __bf16* pf2g = (__bf16*)carve((size_t)N * 128 * 2);
    int*    vox1 = (int*)   carve((size_t)V * 128 * 4);
    __bf16* voxA = (__bf16*)carve((size_t)V * 128 * 2);
    int*    vox2 = (int*)   carve((size_t)V * 256 * 4);
    __bf16* wb3  = (__bf16*)carve(256 * 256 * 2);
    __bf16* wb4  = (__bf16*)carve(256 * 256 * 2);
    __bf16* wbv2 = (__bf16*)carve(256 * 256 * 2);
    __bf16* wbv1 = (__bf16*)carve(128 * 128 * 2);

    (void)hipMemsetAsync(vox1, 0, (size_t)V * 128 * 4, stream);
    (void)hipMemsetAsync(vox2, 0, (size_t)V * 256 * 4, stream);

    k_pack<<<16, 256, 0, stream>>>(w3,  wb3,  256, 256);
    k_pack<<<16, 256, 0, stream>>>(w4,  wb4,  256, 256);
    k_pack<<<16, 256, 0, stream>>>(wv2, wbv2, 256, 256);
    k_pack<<<4,  256, 0, stream>>>(wv1, wbv1, 128, 128);

    k_point_stage1<<<(N + 255) / 256, 256, 0, stream>>>(
        inp, idx, w1x, b1x, w2x, b2x, w1r, b1r, w2r, b2r,
        wax, bax, war, bar, pf2g, vox1, N);

    k_gemm_rows<128, 128, true><<<(V + 31) / 32, 256, 0, stream>>>(
        (const float*)vox1, wbv1, bv1, (void*)voxA, V);

    static_assert(S2_SMEM_BYTES <= 320*1024, "LDS budget");
    (void)hipFuncSetAttribute((const void*)k_stage2_persist,
                              hipFuncAttributeMaxDynamicSharedMemorySize, S2_SMEM_BYTES);
    int pblocks = 512;
    const int nChunks = (N + 31) / 32;
    if (pblocks > nChunks) pblocks = nChunks;
    k_stage2_persist<<<pblocks, 256, S2_SMEM_BYTES, stream>>>(
        pf2g, voxA, idx, wb3, b3, wb4, b4, vox2, N);

    k_gemm_rows<256, 256, false><<<(V + 31) / 32, 256, 0, stream>>>(
        (const float*)vox2, wbv2, bv2, d_out, V);
}
